// ContextQueryAttention_309237646027
// MI455X (gfx1250) — matrix operand order repaired and checked by reading, not compiled, not run on hardware
//
#include <hip/hip_runtime.h>

// ---------------------------------------------------------------------------
// ContextQueryAttention (BiDAF-style trilinear attention) for MI455X/gfx1250.
// All GEMMs via v_wmma_f32_16x16x32_bf16 (wave32). Bm = S1 @ (S2^T c)
// factorization avoids the 256MB Lc x Lc intermediate.
// Round 2: kill accumulator scratch-spills (launch_bounds + compile-time
// leading dims + minimal live B-fragments).
// ---------------------------------------------------------------------------

typedef __attribute__((ext_vector_type(16))) __bf16 v16bf;
typedef __attribute__((ext_vector_type(8)))  float  v8f;

static constexpr int NB  = 16;    // batch
static constexpr int LC  = 2048;  // context length
static constexpr int LQ  = 512;   // question length
static constexpr int DD  = 512;   // feature dim
static constexpr int F4D = 2048;  // 4*D concat features

// round-to-nearest-even f32 -> bf16 (raw bits)
__device__ __forceinline__ unsigned short f2bf(float f) {
  unsigned int u = __builtin_bit_cast(unsigned int, f);
  u += 0x7FFFu + ((u >> 16) & 1u);
  return (unsigned short)(u >> 16);
}

// A-fragment: 16-bit A 16x32 layout. Lane holds row M=lane%16,
// K = [h, h+7] and [h+16, h+23], h = 8*(lane/16).
// Caller passes p = base + row*ld + kb + 8*(lane>>4).
__device__ __forceinline__ v16bf load_a(const unsigned short* p) {
  union { v16bf v; uint4 u[2]; } f;
  f.u[0] = *(const uint4*)(p);
  f.u[1] = *(const uint4*)(p + 16);
  return f.v;
}

// B-fragment: 16-bit B 32x16 layout. Lane holds col N=lane%16,
// K = [h, h+7] and [h+16, h+23], h = 8*(lane/16).
// Caller passes p = Bt_base + n*ld + kb + 8*(lane>>4)  (Bt is N x K row-major).
__device__ __forceinline__ v16bf load_b(const unsigned short* p) {
  union { v16bf v; uint4 u[2]; } f;
  f.u[0] = *(const uint4*)(p);
  f.u[1] = *(const uint4*)(p + 16);
  return f.v;
}

__device__ __forceinline__ v8f wmma_bf16(v16bf a, v16bf b, v8f c) {
  return __builtin_amdgcn_wmma_f32_16x16x32_bf16(
      /*neg_a=*/false, a, /*neg_b=*/false, b,
      /*c_mod=*/(short)0, c, /*reuse_a=*/false, /*reuse_b=*/false);
}

// Generic wave-level NT GEMM: A (M x K row-major, ld = LDA), Bt (N x K
// row-major, ld = LDB), both leading dims compile-time so all tile loads are
// one base pointer + immediate offsets. Wave computes MT*16 x NT*16 at
// (m0, n0). B fragments are loaded per-ni and consumed immediately to keep
// VGPR liveness at acc(8*MT*NT) + a(8*MT) + b(8).
template <int MT, int NT, int LDA, int LDB>
__device__ __forceinline__ void gemm_nt(const unsigned short* __restrict__ A,
                                        const unsigned short* __restrict__ Bt,
                                        int K, int m0, int n0, int lane,
                                        v8f (&acc)[MT][NT]) {
  const v8f zero = {0.f, 0.f, 0.f, 0.f, 0.f, 0.f, 0.f, 0.f};
#pragma unroll
  for (int mi = 0; mi < MT; ++mi)
#pragma unroll
    for (int ni = 0; ni < NT; ++ni) acc[mi][ni] = zero;

  const int r = lane & 15;
  const unsigned short* pa = A  + (size_t)(m0 + r) * LDA + 8  * (lane >> 4);
  const unsigned short* pb = Bt + (size_t)(n0 + r) * LDB + 8 * (lane >> 4);

  for (int kb = 0; kb < K; kb += 32) {
    v16bf a[MT];
#pragma unroll
    for (int mi = 0; mi < MT; ++mi) a[mi] = load_a(pa + mi * 16 * LDA);
    __builtin_prefetch(pa + 128, 0, 3);  // global_prefetch_b8, 4 k-tiles ahead
    __builtin_prefetch(pb + 128, 0, 3);
#pragma unroll
    for (int ni = 0; ni < NT; ++ni) {
      v16bf b = load_b(pb + ni * 16 * LDB);
#pragma unroll
      for (int mi = 0; mi < MT; ++mi)
        acc[mi][ni] = wmma_bf16(a[mi], b, acc[mi][ni]);
    }
    pa += 32;
    pb += 32;
  }
}

// ---------------------------------------------------------------------------
// Prep kernels: bf16 conversions, transposed copies, per-row dot products.
// ---------------------------------------------------------------------------

// c -> cm = bf16(c*wm) (row-major), cT = bf16(c^T) (B,D,Lc), F[:, 0:512] = bf16(c)
__global__ void kPrepC(const float* __restrict__ c, const float* __restrict__ wm,
                       unsigned short* __restrict__ cm, unsigned short* __restrict__ cT,
                       unsigned short* __restrict__ F) {
  const size_t N = (size_t)NB * LC * DD;
  for (size_t idx = (size_t)blockIdx.x * blockDim.x + threadIdx.x; idx < N;
       idx += (size_t)gridDim.x * blockDim.x) {
    int d = (int)(idx % DD);
    size_t bi = idx / DD;            // b*LC + i
    int b = (int)(bi / LC);
    int i = (int)(bi % LC);
    float v = c[idx];
    cm[idx] = f2bf(v * wm[d]);
    cT[((size_t)b * DD + d) * LC + i] = f2bf(v);
    F[bi * F4D + d] = f2bf(v);
  }
}

// q -> qbf = bf16(q) (row-major), qT = bf16(q^T) (B,D,Lq)
__global__ void kPrepQ(const float* __restrict__ q, unsigned short* __restrict__ qbf,
                       unsigned short* __restrict__ qT) {
  const size_t N = (size_t)NB * LQ * DD;
  for (size_t idx = (size_t)blockIdx.x * blockDim.x + threadIdx.x; idx < N;
       idx += (size_t)gridDim.x * blockDim.x) {
    int d = (int)(idx % DD);
    size_t bj = idx / DD;
    int b = (int)(bj / LQ);
    int j = (int)(bj % LQ);
    float v = q[idx];
    qbf[idx] = f2bf(v);
    qT[((size_t)b * DD + d) * LQ + j] = f2bf(v);
  }
}

// wr (4D x D) -> wrT = bf16(wr^T) (D x 4D)
__global__ void kPrepW(const float* __restrict__ wr, unsigned short* __restrict__ wrT) {
  const int N = F4D * DD;
  for (int idx = blockIdx.x * blockDim.x + threadIdx.x; idx < N;
       idx += gridDim.x * blockDim.x) {
    int k = idx / DD, n = idx % DD;
    wrT[(size_t)n * F4D + k] = f2bf(wr[idx]);
  }
}

// out[row] = dot(X[row, 0:512], w)   — one wave32 per row, D=512=32*16
__global__ void kRowDot(const float* __restrict__ X, const float* __restrict__ w,
                        float* __restrict__ out, int rows) {
  int wv = blockIdx.x * (blockDim.x >> 5) + (threadIdx.x >> 5);
  int lane = threadIdx.x & 31;
  if (wv >= rows) return;
  const float* row = X + (size_t)wv * DD;
  float s = 0.f;
#pragma unroll
  for (int t = 0; t < 16; t += 4) {
    float4 v  = *(const float4*)(row + lane * 16 + t);
    float4 ww = *(const float4*)(w + lane * 16 + t);
    s += v.x * ww.x + v.y * ww.y + v.z * ww.z + v.w * ww.w;
  }
#pragma unroll
  for (int off = 16; off > 0; off >>= 1) s += __shfl_xor(s, off, 32);
  if (lane == 0) out[wv] = s;
}

// ---------------------------------------------------------------------------
// S = cm @ q^T + cwc[i] + qwq[j]   (per batch: 2048 x 512, K = 512)
// Block 128 thr = 4 waves, block tile 128x128, wave tile 64x64 (4x4 WMMA).
// ---------------------------------------------------------------------------
__global__ void __launch_bounds__(128, 1)
kS(const unsigned short* __restrict__ cm,
   const unsigned short* __restrict__ qbf,
   const float* __restrict__ cwc, const float* __restrict__ qwq,
   float* __restrict__ S) {
  const int b = blockIdx.y;
  const int nbn = LQ / 128;                       // 4
  const int mblk = blockIdx.x / nbn, nblk = blockIdx.x % nbn;
  const int w = threadIdx.x >> 5, lane = threadIdx.x & 31;
  const int m0 = mblk * 128 + (w >> 1) * 64;
  const int n0 = nblk * 128 + (w & 1) * 64;

  const unsigned short* A  = cm  + (size_t)b * LC * DD;
  const unsigned short* Bt = qbf + (size_t)b * LQ * DD;
  v8f acc[4][4];
  gemm_nt<4, 4, DD, DD>(A, Bt, DD, m0, n0, lane, acc);

  float* Sb = S + (size_t)b * LC * LQ;
#pragma unroll
  for (int mi = 0; mi < 4; ++mi)
#pragma unroll
    for (int r = 0; r < 8; ++r) {
      int m = m0 + 16 * mi + 8 * (lane >> 4) + r;
      float cw = cwc[b * LC + m];
      float* Srow = Sb + (size_t)m * LQ;
#pragma unroll
      for (int ni = 0; ni < 4; ++ni) {
        int n = n0 + 16 * ni + (lane & 15);
        Srow[n] = acc[mi][ni][r] + cw + qwq[b * LQ + n];
      }
    }
}

// ---------------------------------------------------------------------------
// Column softmax (axis=1, length Lc): S1 = softmax_i(S), stored bf16.
// Block (64 j, 4 i-groups); coalesced along j.
// ---------------------------------------------------------------------------
__global__ void kSoftmaxCol(const float* __restrict__ S, unsigned short* __restrict__ S1) {
  const int b = blockIdx.y;
  const int tx = threadIdx.x, ty = threadIdx.y;
  const int j = blockIdx.x * 64 + tx;
  __shared__ float red[4][64];
  __shared__ float colmax[64], colsum[64];
  const float* Sb = S + (size_t)b * LC * LQ;

  float m = -3.4e38f;
  for (int i = ty; i < LC; i += 4) m = fmaxf(m, Sb[(size_t)i * LQ + j]);
  red[ty][tx] = m;
  __syncthreads();
  if (ty == 0)
    colmax[tx] = fmaxf(fmaxf(red[0][tx], red[1][tx]), fmaxf(red[2][tx], red[3][tx]));
  __syncthreads();
  const float mx = colmax[tx];

  float s = 0.f;
  for (int i = ty; i < LC; i += 4) s += expf(Sb[(size_t)i * LQ + j] - mx);
  __syncthreads();
  red[ty][tx] = s;
  __syncthreads();
  if (ty == 0) colsum[tx] = red[0][tx] + red[1][tx] + red[2][tx] + red[3][tx];
  __syncthreads();
  const float inv = 1.f / colsum[tx];

  unsigned short* S1b = S1 + (size_t)b * LC * LQ;
  for (int i = ty; i < LC; i += 4)
    S1b[(size_t)i * LQ + j] = f2bf(expf(Sb[(size_t)i * LQ + j] - mx) * inv);
}

// ---------------------------------------------------------------------------
// Row softmax (axis=2, length Lq=512): one wave per row; writes S2^T
// directly (S2t[b, j, k]) so the T GEMM is NT with contiguous fragments.
// ---------------------------------------------------------------------------
__global__ void kSoftmaxRow(const float* __restrict__ S, unsigned short* __restrict__ S2t) {
  const int w = threadIdx.x >> 5, lane = threadIdx.x & 31;
  const int row = blockIdx.x * 4 + w;  // b*LC + k
  const int b = row / LC, k = row % LC;
  const float* Sr = S + (size_t)row * LQ;

  float v[16];
#pragma unroll
  for (int t = 0; t < 16; t += 4) {
    float4 x = *(const float4*)(Sr + lane * 16 + t);
    v[t] = x.x; v[t + 1] = x.y; v[t + 2] = x.z; v[t + 3] = x.w;
  }
  float m = v[0];
#pragma unroll
  for (int t = 1; t < 16; ++t) m = fmaxf(m, v[t]);
#pragma unroll
  for (int off = 16; off > 0; off >>= 1) m = fmaxf(m, __shfl_xor(m, off, 32));
  float s = 0.f;
#pragma unroll
  for (int t = 0; t < 16; ++t) { v[t] = expf(v[t] - m); s += v[t]; }
#pragma unroll
  for (int off = 16; off > 0; off >>= 1) s += __shfl_xor(s, off, 32);
  const float inv = 1.f / s;

  unsigned short* base = S2t + (size_t)b * LQ * LC + k;
#pragma unroll
  for (int t = 0; t < 16; ++t)
    base[(size_t)(lane * 16 + t) * LC] = f2bf(v[t] * inv);
}

// ---------------------------------------------------------------------------
// T = S2^T @ c  (per batch: Lq x D, K = Lc). A = S2t (Lq x Lc), Bt = cT (D x Lc).
// Output stored transposed: Tt[b, d, j]  (contiguous 16B bf16 stores per lane).
// ---------------------------------------------------------------------------
__global__ void __launch_bounds__(128, 1)
kT(const unsigned short* __restrict__ S2t,
   const unsigned short* __restrict__ cT,
   unsigned short* __restrict__ Tt) {
  const int b = blockIdx.y;
  const int nbn = DD / 128;                       // 4
  const int mblk = blockIdx.x / nbn, nblk = blockIdx.x % nbn;
  const int w = threadIdx.x >> 5, lane = threadIdx.x & 31;
  const int m0 = mblk * 128 + (w >> 1) * 64;      // j
  const int n0 = nblk * 128 + (w & 1) * 64;       // d

  const unsigned short* A  = S2t + (size_t)b * LQ * LC;
  const unsigned short* Bt = cT  + (size_t)b * DD * LC;
  v8f acc[4][4];
  gemm_nt<4, 4, LC, LC>(A, Bt, LC, m0, n0, lane, acc);

  unsigned short* Tb = Tt + (size_t)b * DD * LQ;
#pragma unroll
  for (int mi = 0; mi < 4; ++mi)
#pragma unroll
    for (int ni = 0; ni < 4; ++ni) {
      int n  = n0 + 16 * ni + (lane & 15);
      int mb = m0 + 16 * mi + 8 * (lane >> 4);
      unsigned short tmp[8];
#pragma unroll
      for (int r = 0; r < 8; ++r) tmp[r] = f2bf(acc[mi][ni][r]);
      *(uint4*)(Tb + (size_t)n * LQ + mb) = *(const uint4*)tmp;
    }
}

// ---------------------------------------------------------------------------
// A = S1 @ q and Bm = S1 @ T with shared S1 fragments (K = Lq).
// Bt1 = qT (D x Lq), Bt2 = Tt (D x Lq). Fused epilogue writes
// F[:,512:1024]=A, F[:,1024:1536]=c*A, F[:,1536:2048]=c*Bm (bf16).
// Wave tile 32x64 (2x4) with dual accumulators; 4 waves stacked in M.
// ---------------------------------------------------------------------------
__global__ void __launch_bounds__(128, 1)
kAB(const unsigned short* __restrict__ S1,
    const unsigned short* __restrict__ qT,
    const unsigned short* __restrict__ Tt,
    const float* __restrict__ c,
    unsigned short* __restrict__ F) {
  const int b = blockIdx.y;
  const int nbn = DD / 64;                        // 8
  const int mblk = blockIdx.x / nbn, nblk = blockIdx.x % nbn;
  const int w = threadIdx.x >> 5, lane = threadIdx.x & 31;
  const int m0 = mblk * 128 + w * 32;
  const int n0 = nblk * 64;

  const unsigned short* A  = S1 + (size_t)b * LC * LQ;
  const unsigned short* B1 = qT + (size_t)b * DD * LQ;
  const unsigned short* B2 = Tt + (size_t)b * DD * LQ;

  const v8f zero = {0.f, 0.f, 0.f, 0.f, 0.f, 0.f, 0.f, 0.f};
  v8f acc1[2][4], acc2[2][4];
#pragma unroll
  for (int mi = 0; mi < 2; ++mi)
#pragma unroll
    for (int ni = 0; ni < 4; ++ni) { acc1[mi][ni] = zero; acc2[mi][ni] = zero; }

  const int r = lane & 15;
  const unsigned short* pa  = A  + (size_t)(m0 + r) * LQ + 8  * (lane >> 4);
  const unsigned short* pb1 = B1 + (size_t)(n0 + r) * LQ + 8 * (lane >> 4);
  const unsigned short* pb2 = B2 + (size_t)(n0 + r) * LQ + 8 * (lane >> 4);

  for (int kb = 0; kb < LQ; kb += 32) {
    v16bf a[2];
#pragma unroll
    for (int mi = 0; mi < 2; ++mi) a[mi] = load_a(pa + mi * 16 * LQ);
    __builtin_prefetch(pa + 128, 0, 3);
    __builtin_prefetch(pb1 + 128, 0, 3);
    __builtin_prefetch(pb2 + 128, 0, 3);
#pragma unroll
    for (int ni = 0; ni < 4; ++ni) {
      v16bf b1 = load_b(pb1 + ni * 16 * LQ);
      v16bf b2 = load_b(pb2 + ni * 16 * LQ);
#pragma unroll
      for (int mi = 0; mi < 2; ++mi) {
        acc1[mi][ni] = wmma_bf16(a[mi], b1, acc1[mi][ni]);
        acc2[mi][ni] = wmma_bf16(a[mi], b2, acc2[mi][ni]);
      }
    }
    pa += 32;
    pb1 += 32;
    pb2 += 32;
  }

#pragma unroll
  for (int mi = 0; mi < 2; ++mi)
#pragma unroll
    for (int rr = 0; rr < 8; ++rr) {
      int m = m0 + 16 * mi + 8 * (lane >> 4) + rr;
      const float* crow = c + ((size_t)b * LC + m) * DD;
      unsigned short* Frow = F + ((size_t)b * LC + m) * F4D;
#pragma unroll
      for (int ni = 0; ni < 4; ++ni) {
        int n = n0 + 16 * ni + (lane & 15);
        float av = acc1[mi][ni][rr];
        float bv = acc2[mi][ni][rr];
        float cv = crow[n];
        Frow[512 + n]  = f2bf(av);
        Frow[1024 + n] = f2bf(cv * av);
        Frow[1536 + n] = f2bf(cv * bv);
      }
    }
}

// ---------------------------------------------------------------------------
// out = F @ wr + br.  A = F (32768 x 2048), Bt = wrT (512 x 2048). fp32 out.
// ---------------------------------------------------------------------------
__global__ void __launch_bounds__(128, 1)
kOut(const unsigned short* __restrict__ F,
     const unsigned short* __restrict__ wrT,
     const float* __restrict__ br, float* __restrict__ out) {
  const int nbn = DD / 128;                       // 4
  const int mblk = blockIdx.x / nbn, nblk = blockIdx.x % nbn;
  const int w = threadIdx.x >> 5, lane = threadIdx.x & 31;
  const int m0 = mblk * 128 + (w >> 1) * 64;
  const int n0 = nblk * 128 + (w & 1) * 64;

  v8f acc[4][4];
  gemm_nt<4, 4, F4D, F4D>(F, wrT, F4D, m0, n0, lane, acc);

#pragma unroll
  for (int mi = 0; mi < 4; ++mi)
#pragma unroll
    for (int r = 0; r < 8; ++r) {
      int m = m0 + 16 * mi + 8 * (lane >> 4) + r;
      float* orow = out + (size_t)m * DD;
#pragma unroll
      for (int ni = 0; ni < 4; ++ni) {
        int n = n0 + 16 * ni + (lane & 15);
        orow[n] = acc[mi][ni][r] + br[n];
      }
    }
}

// ---------------------------------------------------------------------------
// Host-side launch
// ---------------------------------------------------------------------------
extern "C" void kernel_launch(void* const* d_in, const int* in_sizes, int n_in,
                              void* d_out, int out_size, void* d_ws, size_t ws_size,
                              hipStream_t stream) {
  const float* c  = (const float*)d_in[0];
  const float* q  = (const float*)d_in[1];
  const float* w0 = (const float*)d_in[2];
  const float* wr = (const float*)d_in[3];
  const float* br = (const float*)d_in[4];
  float* out = (float*)d_out;

  const float* wc = w0;
  const float* wq = w0 + DD;
  const float* wm = w0 + 2 * DD;

  // Workspace layout (all 256B aligned by construction; ~347 MiB total)
  char* p = (char*)d_ws;
  auto take = [&](size_t bytes) { char* r = p; p += (bytes + 255) & ~(size_t)255; return r; };
  float*          S    = (float*)         take((size_t)NB * LC * LQ * 4);   // 64 MiB
  unsigned short* S1   = (unsigned short*)take((size_t)NB * LC * LQ * 2);   // 32 MiB
  unsigned short* S2t  = (unsigned short*)take((size_t)NB * LQ * LC * 2);   // 32 MiB
  unsigned short* cm   = (unsigned short*)take((size_t)NB * LC * DD * 2);   // 32 MiB
  unsigned short* cT   = (unsigned short*)take((size_t)NB * DD * LC * 2);   // 32 MiB
  unsigned short* qbf  = (unsigned short*)take((size_t)NB * LQ * DD * 2);   //  8 MiB
  unsigned short* qT   = (unsigned short*)take((size_t)NB * DD * LQ * 2);   //  8 MiB
  unsigned short* Tt   = (unsigned short*)take((size_t)NB * DD * LQ * 2);   //  8 MiB
  unsigned short* wrT  = (unsigned short*)take((size_t)F4D * DD * 2);       //  2 MiB
  unsigned short* Fm   = (unsigned short*)take((size_t)NB * LC * F4D * 2);  // 128 MiB
  float*          cwc  = (float*)         take((size_t)NB * LC * 4);
  float*          qwq  = (float*)         take((size_t)NB * LQ * 4);

  // 1) prep: bf16 conversions / transposes / row-dot biases
  kPrepC<<<4096, 256, 0, stream>>>(c, wm, cm, cT, Fm);
  kPrepQ<<<2048, 256, 0, stream>>>(q, qbf, qT);
  kPrepW<<<512, 256, 0, stream>>>(wr, wrT);
  kRowDot<<<(NB * LC) / 4, 128, 0, stream>>>(c, wc, cwc, NB * LC);
  kRowDot<<<(NB * LQ) / 4, 128, 0, stream>>>(q, wq, qwq, NB * LQ);

  // 2) S = cm @ q^T + biases
  kS<<<dim3((LC / 128) * (LQ / 128), NB), 128, 0, stream>>>(cm, qbf, cwc, qwq, S);

  // 3) softmaxes
  kSoftmaxCol<<<dim3(LQ / 64, NB), dim3(64, 4), 0, stream>>>(S, S1);
  kSoftmaxRow<<<(NB * LC) / 4, 128, 0, stream>>>(S, S2t);

  // 4) T = S2^T @ c   (stored transposed)
  kT<<<dim3((LQ / 128) * (DD / 128), NB), 128, 0, stream>>>(S2t, cT, Tt);

  // 5) A & Bm + fused feature build
  kAB<<<dim3((LC / 128) * (DD / 64), NB), 128, 0, stream>>>(S1, qT, Tt, c, Fm);

  // 6) out = F @ wr + br
  kOut<<<((NB * LC) / 128) * (DD / 128), 128, 0, stream>>>(Fm, wrT, br, out);
}
